// FuncGNN_64553358459103
// MI455X (gfx1250) — matrix operand order repaired and checked by reading, not compiled, not run on hardware
//
#include <hip/hip_runtime.h>
#include <hip/hip_bf16.h>
#include <math.h>

typedef __attribute__((ext_vector_type(16))) _Float16 v16h;
typedef __attribute__((ext_vector_type(8)))  float    v8f;

#define DD 64

__device__ __forceinline__ float gelu_f(float x) {
    // exact GELU: 0.5*x*(1+erf(x/sqrt(2)))
    return 0.5f * x * (1.0f + erff(x * 0.70710678118654752f));
}

// ---------------------------------------------------------------------------
// Repack a K x M (row-major, f32) weight matrix into WMMA-B register order,
// f16:  P[kchunk][ntile][lane][16 halves], lane = hi*16 + m,
//       half j <-> K = kchunk*32 + (j>>3)*16 + hi*8 + (j&7), N = ntile*16 + m.
// Each lane of the GEMM then loads its 16 halves as 32 contiguous bytes.
// ---------------------------------------------------------------------------
__global__ void repack_b(const float* __restrict__ W, _Float16* __restrict__ P,
                         int M, long total)
{
    long t = (long)blockIdx.x * blockDim.x + threadIdx.x;
    if (t >= total) return;
    int k  = (int)(t / M);
    int nc = (int)(t % M);
    int kc  = k >> 5, kin = k & 31;
    int hi  = (kin >> 3) & 1, j  = (kin & 7) | ((kin >> 4) << 3);
    int nt  = nc >> 4,  m  = nc & 15;
    int lane = hi * 16 + m;
    int MT = M >> 4;
    size_t dst = ((((size_t)kc * MT + nt) * 32) + lane) * 16 + j;
    P[dst] = (_Float16)W[t];
}

// ---------------------------------------------------------------------------
// Generic WMMA GEMM:  C[n x (MT*16)] = act( A[n x K] @ W[K x (MT*16)] + bias )
// A fp32 (row stride lda), W pre-packed f16 (repack_b), C fp32 (row stride ldc).
// One wave per 16-row tile, MT 16-col tiles register-blocked, K chunks of 32.
// ---------------------------------------------------------------------------
template <int ACT, int MT>
__global__ __launch_bounds__(256) void gemm_wmma(
    const float* __restrict__ A, int lda,
    const _Float16* __restrict__ Wp,          // packed [K/32][MT][32][16]
    const float* __restrict__ bias,           // MT*16
    float* __restrict__ C, int ldc,
    int n, int K)
{
    const int lane = threadIdx.x & 31;
    const int wave = threadIdx.x >> 5;
    const long row0 = ((long)blockIdx.x * 8 + wave) * 16;
    if (row0 >= n) return;                    // wave-uniform: EXEC stays all-ones

    const int m  = lane & 15;                 // row-in-tile (A) / col-in-tile (C)
    const int hi = lane >> 4;

    const float* Arow = A + (row0 + m) * (long)lda;
    const _Float16* Wlane = Wp + (size_t)lane * 16;

    v8f acc[MT];
#pragma unroll
    for (int t = 0; t < MT; ++t) {
        v8f z = {0.f,0.f,0.f,0.f,0.f,0.f,0.f,0.f};
        acc[t] = z;
    }

    for (int k0 = 0; k0 < K; k0 += 32) {
        // A-matrix 16x32 f16 layout (ISA 7.12.2):
        // lane<16: halves 0..7 -> K 0..7,  halves 8..15 -> K 16..23
        // lane>=16: halves 0..7 -> K 8..15, halves 8..15 -> K 24..31
        const float* ap = Arow + k0 + hi * 8;
        float4 a0 = *(const float4*)(ap);
        float4 a1 = *(const float4*)(ap + 4);
        float4 a2 = *(const float4*)(ap + 16);
        float4 a3 = *(const float4*)(ap + 20);
        v16h a;
        a[0]  = (_Float16)a0.x; a[1]  = (_Float16)a0.y;
        a[2]  = (_Float16)a0.z; a[3]  = (_Float16)a0.w;
        a[4]  = (_Float16)a1.x; a[5]  = (_Float16)a1.y;
        a[6]  = (_Float16)a1.z; a[7]  = (_Float16)a1.w;
        a[8]  = (_Float16)a2.x; a[9]  = (_Float16)a2.y;
        a[10] = (_Float16)a2.z; a[11] = (_Float16)a2.w;
        a[12] = (_Float16)a3.x; a[13] = (_Float16)a3.y;
        a[14] = (_Float16)a3.z; a[15] = (_Float16)a3.w;

        const _Float16* wk = Wlane + (size_t)(k0 >> 5) * MT * 512;
#pragma unroll
        for (int t = 0; t < MT; ++t) {
            v16h b = *(const v16h*)(wk + (size_t)t * 512);   // 32B contiguous
            acc[t] = __builtin_amdgcn_wmma_f32_16x16x32_f16(
                         false, a, false, b, (short)0, acc[t], false, false);
        }
    }

#pragma unroll
    for (int t = 0; t < MT; ++t) {
        const int col = t * 16 + m;
        const float bv = bias[col];
#pragma unroll
        for (int r = 0; r < 8; ++r) {
            float v = acc[t][r] + bv;
            if (ACT == 1) v = gelu_f(v);
            C[(row0 + hi * 8 + r) * (long)ldc + col] = v;
        }
    }
}

// ---------------------------------------------------------------------------
// Elementwise / scatter / reduction helpers
// ---------------------------------------------------------------------------
__global__ void fill_f(float* __restrict__ p, float v, long n) {
    long t = (long)blockIdx.x * blockDim.x + threadIdx.x;
    if (t < n) p[t] = v;
}

__global__ void deg_count(const int* __restrict__ ei, float* __restrict__ deg, int E) {
    long t = (long)blockIdx.x * blockDim.x + threadIdx.x;
    if (t < E) atomicAdd(&deg[ei[t * 3 + 1]], 1.0f);
}

// agg[dst,:] += (sign?) * src[srcIdx,:]   (one wave per edge, 2 cols per lane)
__global__ void scatter_edges(const int* __restrict__ ei,
                              const float* __restrict__ src, long lds_,
                              float* __restrict__ agg, int E, int useSign)
{
    long t = (long)blockIdx.x * blockDim.x + threadIdx.x;
    if (t >= (long)E * 32) return;
    int e = (int)(t >> 5);
    int c = ((int)t & 31) * 2;
    int s = ei[e * 3 + 0];
    int d = ei[e * 3 + 1];
    float sg = useSign ? (float)ei[e * 3 + 2] : 1.0f;
    float v0 = sg * src[(long)s * lds_ + c];
    float v1 = sg * src[(long)s * lds_ + c + 1];
    atomicAdd(&agg[(long)d * DD + c],     v0);
    atomicAdd(&agg[(long)d * DD + c + 1], v1);
}

// h = (agg + xt)/deg + xprev ; accumulate per-column sum / sumsq
__global__ void hstats(const float* __restrict__ agg, const float* __restrict__ xt,
                       const float* __restrict__ xprev, long ldx,
                       const float* __restrict__ deg, float* __restrict__ h,
                       float* __restrict__ colsum, float* __restrict__ colsq, int n)
{
    const int c  = threadIdx.x & (DD - 1);
    const int rl = threadIdx.x >> 6;  // 0..3
    float s = 0.f, sq = 0.f;
    for (long r = (long)blockIdx.x * 4 + rl; r < n; r += (long)gridDim.x * 4) {
        float v = (agg[r * DD + c] + xt[r * DD + c]) / deg[r] + xprev[r * ldx + c];
        h[r * DD + c] = v;
        s += v; sq += v * v;
    }
    __shared__ float ss[256], sv[256];
    ss[threadIdx.x] = s; sv[threadIdx.x] = sq;
    __syncthreads();
    if (rl == 0) {
#pragma unroll
        for (int i = 1; i < 4; ++i) { s += ss[c + i * DD]; sq += sv[c + i * DD]; }
        atomicAdd(&colsum[c], s);
        atomicAdd(&colsq[c], sq);
    }
}

// fold GraphNorm into per-column scale/shift (64 threads)
__global__ void finalize_norm(const float* __restrict__ colsum, const float* __restrict__ colsq,
                              const float* __restrict__ nw, const float* __restrict__ nb,
                              const float* __restrict__ rsw, const float* __restrict__ rsb,
                              const float* __restrict__ rbw, const float* __restrict__ rbb,
                              const float* __restrict__ rate_b,
                              float* __restrict__ scale, float* __restrict__ shift, float fn)
{
    int c = threadIdx.x;
    float mean = colsum[c] / fn;
    float var  = colsq[c] / fn - mean * mean;
    float inv  = rsqrtf(var + 1e-5f);
    float rate = rate_b[0];
    float gamma = nw[c] + (rsw[c] * rate + rsb[c]);
    float beta  = nb[c] + (rbw[c] * rate + rbb[c]);
    float sc = inv * gamma;
    scale[c] = sc;
    shift[c] = beta - mean * sc;
}

// out[r, off..] = gelu(h*scale + shift), out has row stride ldo
__global__ void apply_norm_gelu(const float* __restrict__ h,
                                const float* __restrict__ scale,
                                const float* __restrict__ shift,
                                float* __restrict__ out, long ldo, long total)
{
    long t = (long)blockIdx.x * blockDim.x + threadIdx.x;
    if (t >= total) return;
    int c = (int)(t & (DD - 1));
    long r = t >> 6;
    out[r * ldo + c] = gelu_f(h[t] * scale[c] + shift[c]);
}

// z = xs(strided) + agg
__global__ void add_strided(const float* __restrict__ xs, long ldx,
                            const float* __restrict__ agg,
                            float* __restrict__ z, long total)
{
    long t = (long)blockIdx.x * blockDim.x + threadIdx.x;
    if (t >= total) return;
    int c = (int)(t & (DD - 1));
    long r = t >> 6;
    z[t] = xs[r * ldx + c] + agg[t];
}

// in-place LayerNorm(128)+ReLU, one wave per row (wave32: 4 elems/lane)
__global__ void ln_relu_kernel(float* __restrict__ X, const float* __restrict__ g,
                               const float* __restrict__ b, int n)
{
    const int lane = threadIdx.x & 31;
    const int wave = threadIdx.x >> 5;
    long r = (long)blockIdx.x * 8 + wave;
    if (r >= n) return;
    float v[4]; float s = 0.f;
#pragma unroll
    for (int j = 0; j < 4; ++j) { v[j] = X[r * 128 + lane + j * 32]; s += v[j]; }
    for (int o = 16; o; o >>= 1) s += __shfl_xor(s, o, 32);
    float mean = s * (1.0f / 128.0f);
    float sq = 0.f;
#pragma unroll
    for (int j = 0; j < 4; ++j) { float d = v[j] - mean; sq += d * d; }
    for (int o = 16; o; o >>= 1) sq += __shfl_xor(sq, o, 32);
    float inv = rsqrtf(sq * (1.0f / 128.0f) + 1e-5f);
#pragma unroll
    for (int j = 0; j < 4; ++j) {
        int col = lane + j * 32;
        float o2 = (v[j] - mean) * inv * g[col] + b[col];
        X[r * 128 + col] = fmaxf(o2, 0.f);
    }
}

// prob = sigmoid(h128 @ W3 + b3), one wave per row
__global__ void prob_kernel(const float* __restrict__ h, const float* __restrict__ W3,
                            const float* __restrict__ b3, float* __restrict__ out, int n)
{
    const int lane = threadIdx.x & 31;
    const int wave = threadIdx.x >> 5;
    long r = (long)blockIdx.x * 8 + wave;
    if (r >= n) return;
    float acc = 0.f;
#pragma unroll
    for (int j = 0; j < 4; ++j) acc += h[r * 128 + lane + j * 32] * W3[lane + j * 32];
    for (int o = 16; o; o >>= 1) acc += __shfl_xor(acc, o, 32);
    if (lane == 0) out[r] = 1.0f / (1.0f + expf(-(acc + b3[0])));
}

// ---------------------------------------------------------------------------
extern "C" void kernel_launch(void* const* d_in, const int* in_sizes, int n_in,
                              void* d_out, int out_size, void* d_ws, size_t ws_size,
                              hipStream_t stream)
{
    const float* init_emb   = (const float*)d_in[0];
    const int*   edge_idx   = (const int*)  d_in[1];
    const float* rate_b     = (const float*)d_in[2];
    const float* sage_lin_W = (const float*)d_in[3];
    const float* sage_lin_b = (const float*)d_in[4];
    const float* sage_nw    = (const float*)d_in[5];
    const float* sage_nb    = (const float*)d_in[6];
    const float* sage_rsw   = (const float*)d_in[7];
    const float* sage_rsb   = (const float*)d_in[8];
    const float* sage_rbw   = (const float*)d_in[9];
    const float* sage_rbb   = (const float*)d_in[10];
    const float* gin_W1     = (const float*)d_in[11];
    const float* gin_b1     = (const float*)d_in[12];
    const float* gin_W2     = (const float*)d_in[13];
    const float* gin_b2     = (const float*)d_in[14];
    const float* fuse_W     = (const float*)d_in[15];
    const float* fuse_b     = (const float*)d_in[16];
    const float* ro_W1      = (const float*)d_in[17];
    const float* ro_b1      = (const float*)d_in[18];
    const float* ro_ln1_g   = (const float*)d_in[19];
    const float* ro_ln1_b   = (const float*)d_in[20];
    const float* ro_W2      = (const float*)d_in[21];
    const float* ro_b2      = (const float*)d_in[22];
    const float* ro_ln2_g   = (const float*)d_in[23];
    const float* ro_ln2_b   = (const float*)d_in[24];
    const float* ro_W3      = (const float*)d_in[25];
    const float* ro_b3      = (const float*)d_in[26];

    const int n = in_sizes[0] / DD;      // 100000
    const int E = in_sizes[1] / 3;       // 1250000

    // ---- workspace carve-up (256B aligned bumps) ----
    char* wp = (char*)d_ws;
    auto alloc = [&](size_t bytes) -> char* {
        char* p = wp;
        wp += (bytes + 255) & ~(size_t)255;
        return p;
    };
    float* deg     = (float*)alloc((size_t)n * 4);
    float* xt      = (float*)alloc((size_t)n * DD * 4);
    float* agg     = (float*)alloc((size_t)n * DD * 4);
    float* hbuf    = (float*)alloc((size_t)n * DD * 4);
    float* xdense  = (float*)alloc((size_t)n * 6 * DD * 4);   // N x 384
    float* h128    = (float*)alloc((size_t)n * 128 * 4);
    float* colsum  = (float*)alloc(256);
    float* colsq   = (float*)alloc(256);
    float* scalev  = (float*)alloc(256);
    float* shiftv  = (float*)alloc(256);
    _Float16* wh_sage = (_Float16*)alloc((size_t)3 * DD * DD * 2);
    _Float16* wh_g1   = (_Float16*)alloc((size_t)3 * DD * DD * 2);
    _Float16* wh_g2   = (_Float16*)alloc((size_t)3 * DD * DD * 2);
    _Float16* wh_fuse = (_Float16*)alloc((size_t)6 * DD * DD * 2);   // 384x64
    _Float16* wh_ro1  = (_Float16*)alloc((size_t)DD * 128 * 2);
    _Float16* wh_ro2  = (_Float16*)alloc((size_t)128 * 128 * 2);
    (void)ws_size; (void)n_in; (void)out_size;

    const int T = 256;
    const long nD = (long)n * DD;
    const int blkND   = (int)((nD + T - 1) / T);
    const int blkRows = (n + 127) / 128;                  // gemm: 128 rows/block
    const int blkWaveRow = (n + 7) / 8;                   // 1 wave per row
    const long scTot = (long)E * 32;
    const int blkScat = (int)((scTot + T - 1) / T);

    // degree = 1 + indegree
    fill_f<<<(n + T - 1) / T, T, 0, stream>>>(deg, 1.0f, n);
    deg_count<<<(E + T - 1) / T, T, 0, stream>>>(edge_idx, deg, E);

    // f16 weight staging in WMMA-B register order
    for (int i = 0; i < 3; ++i) {
        repack_b<<<(DD * DD + T - 1) / T, T, 0, stream>>>(
            sage_lin_W + (size_t)i * DD * DD, wh_sage + (size_t)i * DD * DD, DD, DD * DD);
        repack_b<<<(DD * DD + T - 1) / T, T, 0, stream>>>(
            gin_W1 + (size_t)i * DD * DD, wh_g1 + (size_t)i * DD * DD, DD, DD * DD);
        repack_b<<<(DD * DD + T - 1) / T, T, 0, stream>>>(
            gin_W2 + (size_t)i * DD * DD, wh_g2 + (size_t)i * DD * DD, DD, DD * DD);
    }
    repack_b<<<(6 * DD * DD + T - 1) / T, T, 0, stream>>>(fuse_W, wh_fuse, DD, 6 * DD * DD);
    repack_b<<<(DD * 128 + T - 1) / T, T, 0, stream>>>(ro_W1, wh_ro1, 128, DD * 128);
    repack_b<<<(128 * 128 + T - 1) / T, T, 0, stream>>>(ro_W2, wh_ro2, 128, 128 * 128);

    const float* xptr = init_emb;
    int ldx = DD;
    for (int i = 0; i < 3; ++i) {
        // xt = x @ W_sage[i] + b
        gemm_wmma<0, 4><<<blkRows, T, 0, stream>>>(xptr, ldx, wh_sage + (size_t)i * DD * DD,
                                                   sage_lin_b + i * DD, xt, DD, n, DD);
        // agg = segsum(sign * xt[src] -> dst)
        fill_f<<<blkND, T, 0, stream>>>(agg, 0.0f, nD);
        scatter_edges<<<blkScat, T, 0, stream>>>(edge_idx, xt, DD, agg, E, 1);
        // h = (agg+xt)/deg + x ; column stats
        fill_f<<<1, 64, 0, stream>>>(colsum, 0.0f, 64);
        fill_f<<<1, 64, 0, stream>>>(colsq, 0.0f, 64);
        hstats<<<512, T, 0, stream>>>(agg, xt, xptr, ldx, deg, hbuf, colsum, colsq, n);
        finalize_norm<<<1, 64, 0, stream>>>(colsum, colsq,
                                            sage_nw + i * DD, sage_nb + i * DD,
                                            sage_rsw + i * DD, sage_rsb + i * DD,
                                            sage_rbw + i * DD, sage_rbb + i * DD,
                                            rate_b, scalev, shiftv, (float)n);
        // x = gelu(graphnorm(h))  -> xdense strip 2i
        float* xs = xdense + (size_t)(2 * i) * DD;
        apply_norm_gelu<<<blkND, T, 0, stream>>>(hbuf, scalev, shiftv, xs, 6 * DD, nD);
        // agg2 = segsum(x[src] -> dst) ; z = x + agg2
        fill_f<<<blkND, T, 0, stream>>>(agg, 0.0f, nD);
        scatter_edges<<<blkScat, T, 0, stream>>>(edge_idx, xs, 6 * DD, agg, E, 0);
        add_strided<<<blkND, T, 0, stream>>>(xs, 6 * DD, agg, hbuf, nD);
        // x = gelu(z@W1+b1)@W2+b2  -> xdense strip 2i+1
        gemm_wmma<1, 4><<<blkRows, T, 0, stream>>>(hbuf, DD, wh_g1 + (size_t)i * DD * DD,
                                                   gin_b1 + i * DD, xt, DD, n, DD);
        float* xo = xdense + (size_t)(2 * i + 1) * DD;
        gemm_wmma<0, 4><<<blkRows, T, 0, stream>>>(xt, DD, wh_g2 + (size_t)i * DD * DD,
                                                   gin_b2 + i * DD, xo, 6 * DD, n, DD);
        xptr = xo;
        ldx = 6 * DD;
    }

    // x_final = x_dense @ fuse_W + fuse_b  -> d_out[0 : n*64]
    float* x_final = (float*)d_out;
    gemm_wmma<0, 4><<<blkRows, T, 0, stream>>>(xdense, 6 * DD, wh_fuse, fuse_b,
                                               x_final, DD, n, 6 * DD);

    // readout MLP
    gemm_wmma<0, 8><<<blkRows, T, 0, stream>>>(x_final, DD, wh_ro1, ro_b1, h128, 128, n, DD);
    ln_relu_kernel<<<blkWaveRow, T, 0, stream>>>(h128, ro_ln1_g, ro_ln1_b, n);
    gemm_wmma<0, 8><<<blkRows, T, 0, stream>>>(h128, 128, wh_ro2, ro_b2, h128, 128, n, 128);
    ln_relu_kernel<<<blkWaveRow, T, 0, stream>>>(h128, ro_ln2_g, ro_ln2_b, n);
    prob_kernel<<<blkWaveRow, T, 0, stream>>>(h128, ro_W3, ro_b3, x_final + (size_t)n * DD, n);
}
